// ScaledDotProductAttention_32452772889194
// MI455X (gfx1250) — matrix operand order repaired and checked by reading, not compiled, not run on hardware
//
#include <hip/hip_runtime.h>

typedef __attribute__((ext_vector_type(16))) _Float16 v16h;
typedef __attribute__((ext_vector_type(8)))  _Float16 v8h;
typedef __attribute__((ext_vector_type(8)))  float    v8f;

#define L_CTX 2048
#define DHEAD 64
#define BH    32
#define KT    64                 // keys per iteration
#define NTILE (L_CTX / KT)       // 32 iterations
// 1/sqrt(64) * log2(e): softmax computed in base-2 (exactly equivalent)
#define QSCALE 0.18033688011112042f

__device__ __forceinline__ v8f wmma_f16(v16h a, v16h b, v8f c) {
  // D(16x16 f32) = A(16x32 f16) * B(32x16 f16) + C
  return __builtin_amdgcn_wmma_f32_16x16x32_f16(false, a, false, b, (short)0, c,
                                                false, false);
}

// ---- DPP16 cross-lane helpers (within each 16-lane row; wave32 = 2 rows) ----
template <int CTRL>
__device__ __forceinline__ float dppf(float x) {
  const int xi = __builtin_bit_cast(int, x);
  return __builtin_bit_cast(
      float, __builtin_amdgcn_update_dpp(xi, xi, CTRL, 0xF, 0xF, true));
}
__device__ __forceinline__ float red_max16(float x) {
  x = fmaxf(x, dppf<0xB1>(x));   // quad_perm [1,0,3,2]  (xor 1)
  x = fmaxf(x, dppf<0x4E>(x));   // quad_perm [2,3,0,1]  (xor 2)
  x = fmaxf(x, dppf<0x141>(x));  // row_half_mirror      (combine quads)
  x = fmaxf(x, dppf<0x140>(x));  // row_mirror           (combine octets)
  return x;
}
__device__ __forceinline__ float red_sum16(float x) {
  x += dppf<0xB1>(x);
  x += dppf<0x4E>(x);
  x += dppf<0x141>(x);
  x += dppf<0x140>(x);
  return x;
}

// ---- CDNA5 async copy: 8KB tile, 16B per lane (ASYNCcnt tracked) ----
__device__ __forceinline__ void async_tile8k(const _Float16* __restrict__ src,
                                             const _Float16* dst_lds, int tid) {
#pragma unroll
  for (int c = 0; c < 4; ++c) {
    const unsigned boff = (unsigned)((tid + c * 128) * 16);
    const unsigned loff =
        (unsigned)(unsigned long long)(const void*)dst_lds + boff;
    asm volatile("global_load_async_to_lds_b128 %0, %1, %2"
                 :
                 : "v"(loff), "v"(boff), "s"(src)
                 : "memory");
  }
}
__device__ __forceinline__ void wait_async0() {
  asm volatile("s_wait_asynccnt 0x0" ::: "memory");
}

__global__ void zero_loss_kernel(float* loss) { *loss = 0.0f; }

// loss += B*H * sum(mask^2)
__global__ void masksq_kernel(const float* __restrict__ m, float* __restrict__ loss,
                              int n, float mult) {
  __shared__ float red[256];
  float s = 0.0f;
  for (int i = blockIdx.x * blockDim.x + threadIdx.x; i < n;
       i += gridDim.x * blockDim.x) {
    const float v = m[i];
    s += v * v;
  }
  red[threadIdx.x] = s;
  __syncthreads();
  for (int st = 128; st > 0; st >>= 1) {
    if ((int)threadIdx.x < st) red[threadIdx.x] += red[threadIdx.x + st];
    __syncthreads();
  }
  if (threadIdx.x == 0) atomicAdd(loss, red[0] * mult);
}

// One-shot f32->f16 pre-pass: K16 row-major, Vt16 transposed per 64-key tile
// (layouts match the LDS tiles exactly -> contiguous 8KB async copies).
__global__ __launch_bounds__(256) void cvt_kernel(
    const float* __restrict__ K, const float* __restrict__ V,
    _Float16* __restrict__ K16, _Float16* __restrict__ Vt16) {
  const size_t i = (size_t)blockIdx.x * 256 + threadIdx.x;
  if (i >= (size_t)BH * L_CTX * DHEAD) return;
  K16[i] = (_Float16)K[i];
  const int d    = (int)(i & (DHEAD - 1));
  const size_t r = i >> 6;                 // bh*L + key
  const int key  = (int)(r & (L_CTX - 1));
  const int bh   = (int)(r >> 11);
  const int jt   = key >> 6, kl = key & (KT - 1);
  Vt16[((size_t)(bh * NTILE + jt) * (DHEAD * KT)) + d * KT + kl] = (_Float16)V[i];
}

// Flash-attention forward + (attn - mask)^2 partial loss.
// Block: 128 threads = 4 waves. Each wave: 16 q-rows. Block: 64 q-rows.
template <bool ASYNC>
__global__ __launch_bounds__(128) void attn_fwd_kernel(
    const float* __restrict__ Q, const float* __restrict__ K,
    const float* __restrict__ V, const float* __restrict__ M,
    const _Float16* __restrict__ K16, const _Float16* __restrict__ Vt16,
    float* __restrict__ ctx, float* __restrict__ loss) {
  const int bh    = blockIdx.y;
  const int qbase = blockIdx.x * 64;
  const int tid   = threadIdx.x;
  const int wave  = tid >> 5;
  const int lane  = tid & 31;
  const int hh    = lane >> 4;  // lane half (0/1)
  const int cl    = lane & 15;  // column / A-row within half

  // LDS: double-buffered 64x64 K tiles (2x8KB) + V^T tiles (2x8KB),
  //      Q staging 64x64 f16 (8KB) reused as per-wave P buffers (16x64 each)
  __shared__ __align__(32) _Float16 bufK [2][KT * DHEAD];
  __shared__ __align__(32) _Float16 bufVt[2][DHEAD * KT];
  __shared__ __align__(32) _Float16 ldsQP[64 * DHEAD];

  const size_t base = (size_t)bh * L_CTX * DHEAD;

  if (ASYNC) {  // prefetch tile 0 while we stage Q
    async_tile8k(K16 + base, &bufK[0][0], tid);
    async_tile8k(Vt16 + (size_t)(bh * NTILE) * (DHEAD * KT), &bufVt[0][0], tid);
  }

  // ---- stage Q tile (64 rows) into LDS as f16, pre-scaled ----
  {
    const float* Qg = Q + base + (size_t)qbase * DHEAD;
    for (int i = tid; i < 64 * 16; i += 128) {
      const int r  = i >> 4;
      const int c4 = (i & 15) << 2;
      const float4 q4 = *(const float4*)(Qg + r * DHEAD + c4);
      _Float16* qd = &ldsQP[r * DHEAD + c4];
      qd[0] = (_Float16)(q4.x * QSCALE);
      qd[1] = (_Float16)(q4.y * QSCALE);
      qd[2] = (_Float16)(q4.z * QSCALE);
      qd[3] = (_Float16)(q4.w * QSCALE);
    }
  }
  __syncthreads();

  // ---- build Q A-matrix registers (16x32 f16 x2 halves of D=64) ----
  // A layout as loaded here (contiguous k, same order as B): lane row M = lane&15,
  // element e -> K = 16*(lane>>4) + e
  v16h aQ[2];
  {
    const _Float16* qp = &ldsQP[(wave * 16 + cl) * DHEAD];
#pragma unroll
    for (int dh = 0; dh < 2; ++dh) {
      const v8h lo = *(const v8h*)(qp + dh * 32 + 16 * hh);
      const v8h hi = *(const v8h*)(qp + dh * 32 + 16 * hh + 8);
      v16h a;
#pragma unroll
      for (int e = 0; e < 8; ++e) { a[e] = lo[e]; a[8 + e] = hi[e]; }
      aQ[dh] = a;
    }
  }
  __syncthreads();  // ldsQP now reusable as P buffers

  _Float16* Pw = &ldsQP[wave * (16 * KT)];

  // Per-row online-softmax state; lsum/s2sum/smsum are per-lane partials
  float mstat[8], lsum[8], s2sum[8], smsum[8];
  v8f   O[4];
#pragma unroll
  for (int r = 0; r < 8; ++r) {
    mstat[r] = -1e30f; lsum[r] = 0.f; s2sum[r] = 0.f; smsum[r] = 0.f;
  }
#pragma unroll
  for (int n = 0; n < 4; ++n) O[n] = (v8f){};

  int cur = 0;
  for (int jt = 0; jt < NTILE; ++jt) {
    const int jbase = jt * KT;
    const _Float16* kb;
    const _Float16* vb;
    if (ASYNC) {
      wait_async0();     // this wave's pending copies for tile jt done
      __syncthreads();   // all waves' copies done; prev-buffer reads done
      if (jt + 1 < NTILE) {
        async_tile8k(K16 + base + (size_t)(jbase + KT) * DHEAD,
                     &bufK[cur ^ 1][0], tid);
        async_tile8k(Vt16 + (size_t)(bh * NTILE + jt + 1) * (DHEAD * KT),
                     &bufVt[cur ^ 1][0], tid);
      }
      kb = &bufK[cur][0];
      vb = &bufVt[cur][0];
    } else {
      __syncthreads();
      const float* Kg = K + base + (size_t)jbase * DHEAD;
      const float* Vg = V + base + (size_t)jbase * DHEAD;
      for (int i = tid; i < KT * 16; i += 128) {
        const int r  = i >> 4;
        const int c4 = (i & 15) << 2;
        const float4 k4 = *(const float4*)(Kg + r * DHEAD + c4);
        _Float16* kd = &bufK[0][r * DHEAD + c4];
        kd[0] = (_Float16)k4.x; kd[1] = (_Float16)k4.y;
        kd[2] = (_Float16)k4.z; kd[3] = (_Float16)k4.w;
        const float4 v4 = *(const float4*)(Vg + r * DHEAD + c4);
        bufVt[0][(c4 + 0) * KT + r] = (_Float16)v4.x;
        bufVt[0][(c4 + 1) * KT + r] = (_Float16)v4.y;
        bufVt[0][(c4 + 2) * KT + r] = (_Float16)v4.z;
        bufVt[0][(c4 + 3) * KT + r] = (_Float16)v4.w;
      }
      __syncthreads();
      kb = &bufK[0][0];
      vb = &bufVt[0][0];
    }

    // ---- S = (Q*qscale) * K^T : four 16x16 tiles (64 keys) ----
    // B layout (32x16): lane col N = lane&15, element e -> K = 16*(lane>>4)+e
    v8f s[4];
#pragma unroll
    for (int t = 0; t < 4; ++t) {
      const _Float16* kp = kb + (t * 16 + cl) * DHEAD + 16 * hh;
      const v16h b0 = *(const v16h*)(kp);
      const v16h b1 = *(const v16h*)(kp + 32);
      v8f c = (v8f){};
      c = wmma_f16(aQ[0], b0, c);
      c = wmma_f16(aQ[1], b1, c);
      s[t] = c;
    }

    // ---- online softmax update, base-2 (C layout: row = r+8*hh, col = cl) --
    const float* mrow = M + (size_t)(qbase + wave * 16) * L_CTX + jbase + cl;
#pragma unroll
    for (int r = 0; r < 8; ++r) {
      const float v0 = s[0][r], v1 = s[1][r], v2 = s[2][r], v3 = s[3][r];
      const float mx =
          red_max16(fmaxf(fmaxf(v0, v1), fmaxf(v2, v3)));
      const float mnew  = fmaxf(mstat[r], mx);
      const float alpha = __builtin_amdgcn_exp2f(mstat[r] - mnew);
      mstat[r] = mnew;
      const float p0 = __builtin_amdgcn_exp2f(v0 - mnew);
      const float p1 = __builtin_amdgcn_exp2f(v1 - mnew);
      const float p2 = __builtin_amdgcn_exp2f(v2 - mnew);
      const float p3 = __builtin_amdgcn_exp2f(v3 - mnew);
      const size_t moff = (size_t)(r + 8 * hh) * L_CTX;
      const float mk0 = mrow[moff];
      const float mk1 = mrow[moff + 16];
      const float mk2 = mrow[moff + 32];
      const float mk3 = mrow[moff + 48];
      lsum[r]  = lsum[r] * alpha + ((p0 + p1) + (p2 + p3));
      s2sum[r] = s2sum[r] * alpha * alpha +
                 ((p0 * p0 + p1 * p1) + (p2 * p2 + p3 * p3));
      smsum[r] = smsum[r] * alpha +
                 ((p0 * mk0 + p1 * mk1) + (p2 * mk2 + p3 * mk3));
#pragma unroll
      for (int n = 0; n < 4; ++n) O[n][r] *= alpha;
      // stash P (C layout -> per-wave LDS, f16)
      _Float16* pr = Pw + (r + 8 * hh) * KT + cl;
      pr[0]  = (_Float16)p0;
      pr[16] = (_Float16)p1;
      pr[32] = (_Float16)p2;
      pr[48] = (_Float16)p3;
    }
    __syncthreads();

    // ---- O += P * V : re-read P in A layout, V^T gives contiguous B ----
    v16h pa[2];
    {
      const _Float16* pp = Pw + cl * KT;
#pragma unroll
      for (int kc = 0; kc < 2; ++kc) {
        const v8h lo = *(const v8h*)(pp + kc * 32 + 16 * hh);
        const v8h hi = *(const v8h*)(pp + kc * 32 + 16 * hh + 8);
        v16h a;
#pragma unroll
        for (int e = 0; e < 8; ++e) { a[e] = lo[e]; a[8 + e] = hi[e]; }
        pa[kc] = a;
      }
    }
#pragma unroll
    for (int n = 0; n < 4; ++n) {
      const _Float16* vp = vb + (n * 16 + cl) * KT + 16 * hh;
      const v16h bv0 = *(const v16h*)(vp);
      const v16h bv1 = *(const v16h*)(vp + 32);
      O[n] = wmma_f16(pa[0], bv0, O[n]);
      O[n] = wmma_f16(pa[1], bv1, O[n]);
    }
    if (ASYNC) cur ^= 1;
  }

  // ---- final cross-lane reduction of row stats (DPP butterflies) ----
#pragma unroll
  for (int r = 0; r < 8; ++r) {
    lsum[r]  = red_sum16(lsum[r]);
    s2sum[r] = red_sum16(s2sum[r]);
    smsum[r] = red_sum16(smsum[r]);
  }

  // ---- write context (normalized) + loss partial ----
  float* Cg = ctx + base;
  float lpart = 0.0f;
#pragma unroll
  for (int r = 0; r < 8; ++r) {
    const float invl = 1.0f / lsum[r];
    const int qrow = qbase + wave * 16 + r + 8 * hh;
#pragma unroll
    for (int n = 0; n < 4; ++n)
      Cg[(size_t)qrow * DHEAD + n * 16 + cl] = O[n][r] * invl;
    lpart += s2sum[r] * invl * invl - 2.0f * smsum[r] * invl;
  }
  if (cl == 0) atomicAdd(loss, lpart);  // lanes 0 & 16 cover rows 0-7 / 8-15
}

extern "C" void kernel_launch(void* const* d_in, const int* in_sizes, int n_in,
                              void* d_out, int out_size, void* d_ws, size_t ws_size,
                              hipStream_t stream) {
  const float* Q = (const float*)d_in[0];
  const float* K = (const float*)d_in[1];
  const float* V = (const float*)d_in[2];
  const float* M = (const float*)d_in[3];
  float* ctx  = (float*)d_out;
  float* loss = ctx + (size_t)BH * L_CTX * DHEAD;  // tuple: (context, loss)

  zero_loss_kernel<<<1, 1, 0, stream>>>(loss);
  masksq_kernel<<<1024, 256, 0, stream>>>(M, loss, L_CTX * L_CTX, (float)BH);

  const size_t nElem  = (size_t)BH * L_CTX * DHEAD;
  const size_t needWS = nElem * 2 * sizeof(_Float16);  // K16 + Vt16 = 16 MB
  dim3 grid(L_CTX / 64, BH);

  if (ws_size >= needWS) {
    _Float16* K16  = (_Float16*)d_ws;
    _Float16* Vt16 = K16 + nElem;
    cvt_kernel<<<(int)(nElem / 256), 256, 0, stream>>>(K, V, K16, Vt16);
    attn_fwd_kernel<true><<<grid, 128, 0, stream>>>(Q, K, V, M, K16, Vt16, ctx,
                                                    loss);
  } else {
    attn_fwd_kernel<false><<<grid, 128, 0, stream>>>(Q, K, V, M, nullptr,
                                                     nullptr, ctx, loss);
  }
}
